// TMSA_53102975647995
// MI455X (gfx1250) — matrix operand order repaired and checked by reading, not compiled, not run on hardware
//
#include <hip/hip_runtime.h>
#include <hip/hip_bf16.h>

typedef __attribute__((ext_vector_type(16))) _Float16 v16h;
typedef __attribute__((ext_vector_type(8)))  _Float16 v8h;
typedef __attribute__((ext_vector_type(8)))  float    v8f;

#define CDIM   192
#define NHEAD  6
#define HDIM   32
#define NTOK   128
#define NWIN_B 192      // windows per batch item
#define BWIN   768      // total windows
#define TTOK   98304    // total tokens
#define HID2   384
#define QKVC   576

#define WMMA_F32(a,b,c) __builtin_amdgcn_wmma_f32_16x16x32_f16(false,(a),false,(b),(short)0,(c),false,false)

// CDNA5 §7.12.2 f16 fragment lane maps reduce to contiguous runs:
//  A 16x32 : lane row m=lane&15, kh=lane>>4 ; e0-7 -> k=[8kh,8kh+8), e8-15 -> k=[16+8kh,24+8kh)
//  B 32x16 : lane col n=lane&15, kh=lane>>4 ; e0-7 -> k=[8kh,8kh+8), e8-15 -> k=[16+8kh,24+8kh)
__device__ __forceinline__ v16h fragA_ptr(const _Float16* rowBase, int kh){
  v8h lo = *reinterpret_cast<const v8h*>(rowBase + (kh << 3));
  v8h hi = *reinterpret_cast<const v8h*>(rowBase + 16 + (kh << 3));
  return __builtin_shufflevector(lo, hi, 0,1,2,3,4,5,6,7,8,9,10,11,12,13,14,15);
}
__device__ __forceinline__ v16h fragB_ptr(const _Float16* rowBase, int kh){
  v8h lo = *reinterpret_cast<const v8h*>(rowBase + (kh << 3));
  v8h hi = *reinterpret_cast<const v8h*>(rowBase + 16 + (kh << 3));
  return __builtin_shufflevector(lo, hi, 0,1,2,3,4,5,6,7,8,9,10,11,12,13,14,15);
}

// ============================ prep kernels ============================

// LN1 + roll(-SS) + window partition; one wave per token
__global__ __launch_bounds__(256) void k_prep1(const float* __restrict__ x,
    const float* __restrict__ pos, const float* __restrict__ lnw, const float* __restrict__ lnb,
    _Float16* __restrict__ XW, _Float16* __restrict__ XWPB)
{
  int token = blockIdx.x * 8 + (threadIdx.x >> 5);
  int lane = threadIdx.x & 31;
  int wi = token >> 7, tk = token & 127;
  int b = wi / NWIN_B, rem = wi % NWIN_B;
  int d  = ((rem >> 6) * 2 + (tk >> 6) + 1) % 6;
  int hh = ((((rem >> 3) & 7) << 3) + ((tk >> 3) & 7) + 4) & 63;
  int ww = (((rem & 7) << 3) + (tk & 7) + 4) & 63;
  const float* src = x + ((((long)b * 6 + d) * 64 + hh) * 64 + ww) * CDIM;
  float v[6]; float s = 0.f;
#pragma unroll
  for (int i = 0; i < 6; ++i){ v[i] = src[lane * 6 + i]; s += v[i]; }
#pragma unroll
  for (int m = 1; m < 32; m <<= 1) s += __shfl_xor(s, m);
  float mu = s * (1.f / CDIM);
  float vs = 0.f;
#pragma unroll
  for (int i = 0; i < 6; ++i){ float dd = v[i] - mu; vs += dd * dd; }
#pragma unroll
  for (int m = 1; m < 32; m <<= 1) vs += __shfl_xor(vs, m);
  float rstd = rsqrtf(vs * (1.f / CDIM) + 1e-5f);
  long base = (long)token * CDIM;
#pragma unroll
  for (int i = 0; i < 6; ++i){
    int c = lane * 6 + i;
    float y = (v[i] - mu) * rstd * lnw[c] + lnb[c];
    XW[base + c]   = (_Float16)y;
    XWPB[base + c] = (_Float16)(y + pos[tk * CDIM + c]);
  }
}

// LN2 on residual tensor (linear layout)
__global__ __launch_bounds__(256) void k_prep2(const float* __restrict__ X1,
    const float* __restrict__ lnw, const float* __restrict__ lnb, _Float16* __restrict__ H2)
{
  int token = blockIdx.x * 8 + (threadIdx.x >> 5);
  int lane = threadIdx.x & 31;
  const float* src = X1 + (long)token * CDIM;
  float v[6]; float s = 0.f;
#pragma unroll
  for (int i = 0; i < 6; ++i){ v[i] = src[lane * 6 + i]; s += v[i]; }
#pragma unroll
  for (int m = 1; m < 32; m <<= 1) s += __shfl_xor(s, m);
  float mu = s * (1.f / CDIM);
  float vs = 0.f;
#pragma unroll
  for (int i = 0; i < 6; ++i){ float dd = v[i] - mu; vs += dd * dd; }
#pragma unroll
  for (int m = 1; m < 32; m <<= 1) vs += __shfl_xor(vs, m);
  float rstd = rsqrtf(vs * (1.f / CDIM) + 1e-5f);
  long base = (long)token * CDIM;
#pragma unroll
  for (int i = 0; i < 6; ++i){
    int c = lane * 6 + i;
    H2[base + c] = (_Float16)((v[i] - mu) * rstd * lnw[c] + lnb[c]);
  }
}

// dense relative position bias (NH,128,128)
__global__ __launch_bounds__(256) void k_rpb(const float* __restrict__ tbl,
    const int* __restrict__ rpi, float* __restrict__ rpbd)
{
  int idx = blockIdx.x * 256 + threadIdx.x;
  int h = idx >> 14, ij = idx & 16383;
  rpbd[idx] = tbl[rpi[ij] * NHEAD + h];
}

// straight f32 -> f16 convert, weight stays in natural (N,K) layout
__global__ __launch_bounds__(256) void k_conv(const float* __restrict__ w,
    _Float16* __restrict__ o, int nel)
{
  int idx = blockIdx.x * 256 + threadIdx.x;
  if (idx >= nel) return;
  o[idx] = (_Float16)w[idx];
}

// ============================ GEMM kernels ============================
// 256 threads = 8 waves arranged 4(M)x2(N); block tile 128x64; wave tile 32x32.
// A staged [m][k] (stride 40), B staged [n][k] (stride 40) straight from (N,K) weights.

__global__ __launch_bounds__(256) void k_gemm_qkv(const _Float16* __restrict__ A,
    const _Float16* __restrict__ Bw, const float* __restrict__ bias,
    _Float16* __restrict__ Out, int K, int Nc, float qscale)
{
  __shared__ _Float16 Ash[128 * 40];
  __shared__ _Float16 Bsh[64 * 40];
  int t = threadIdx.x, lane = t & 31, wv = t >> 5;
  int wm = wv >> 1, wn = wv & 1;
  int n = lane & 15, kh = lane >> 4, mg = lane >> 4;
  int mBase = blockIdx.x * 128, nBase = blockIdx.y * 64;
  v8f acc[2][2] = {};
  for (int k0 = 0; k0 < K; k0 += 32){
    int ar = t >> 1, ac = (t & 1) << 4;
    const uint4* as = reinterpret_cast<const uint4*>(A + (long)(mBase + ar) * K + k0 + ac);
    if (k0 + 32 < K) __builtin_prefetch(as + 4, 0, 0);
    uint4 a0 = as[0], a1 = as[1];
    uint4* ad = reinterpret_cast<uint4*>(Ash + ar * 40 + ac);
    ad[0] = a0; ad[1] = a1;
    int bn = t >> 2, bk = (t & 3) << 3;
    uint4 bv = *reinterpret_cast<const uint4*>(Bw + (long)(nBase + bn) * K + k0 + bk);
    *reinterpret_cast<uint4*>(Bsh + bn * 40 + bk) = bv;
    __syncthreads();
    v16h aF0 = fragA_ptr(Ash + (wm * 32 + n) * 40, kh);
    v16h aF1 = fragA_ptr(Ash + (wm * 32 + 16 + n) * 40, kh);
    v16h bF0 = fragB_ptr(Bsh + (wn * 32 + n) * 40, kh);
    v16h bF1 = fragB_ptr(Bsh + (wn * 32 + 16 + n) * 40, kh);
    acc[0][0] = WMMA_F32(aF0, bF0, acc[0][0]);
    acc[0][1] = WMMA_F32(aF0, bF1, acc[0][1]);
    acc[1][0] = WMMA_F32(aF1, bF0, acc[1][0]);
    acc[1][1] = WMMA_F32(aF1, bF1, acc[1][1]);
    __syncthreads();
  }
#pragma unroll
  for (int i = 0; i < 2; ++i)
#pragma unroll
    for (int jn = 0; jn < 2; ++jn){
      int col = nBase + wn * 32 + jn * 16 + n;
      float bval = bias[col];
      float sc = (col < CDIM) ? qscale : 1.f;     // scale q columns
#pragma unroll
      for (int r = 0; r < 8; ++r){
        int row = mBase + wm * 32 + i * 16 + r + (mg << 3);
        Out[(long)row * Nc + col] = (_Float16)((acc[i][jn][r] + bval) * sc);
      }
    }
}

// proj GEMM with fused window-reverse + roll(+SS) + residual add
__global__ __launch_bounds__(256) void k_gemm_proj(const _Float16* __restrict__ A,
    const _Float16* __restrict__ Bw, const float* __restrict__ bias,
    const float* __restrict__ xin, float* __restrict__ X1)
{
  const int K = 2 * CDIM;
  __shared__ _Float16 Ash[128 * 40];
  __shared__ _Float16 Bsh[64 * 40];
  int t = threadIdx.x, lane = t & 31, wv = t >> 5;
  int wm = wv >> 1, wn = wv & 1;
  int n = lane & 15, kh = lane >> 4, mg = lane >> 4;
  int mBase = blockIdx.x * 128, nBase = blockIdx.y * 64;
  v8f acc[2][2] = {};
  for (int k0 = 0; k0 < K; k0 += 32){
    int ar = t >> 1, ac = (t & 1) << 4;
    const uint4* as = reinterpret_cast<const uint4*>(A + (long)(mBase + ar) * K + k0 + ac);
    if (k0 + 32 < K) __builtin_prefetch(as + 4, 0, 0);
    uint4 a0 = as[0], a1 = as[1];
    uint4* ad = reinterpret_cast<uint4*>(Ash + ar * 40 + ac);
    ad[0] = a0; ad[1] = a1;
    int bn = t >> 2, bk = (t & 3) << 3;
    uint4 bv = *reinterpret_cast<const uint4*>(Bw + (long)(nBase + bn) * K + k0 + bk);
    *reinterpret_cast<uint4*>(Bsh + bn * 40 + bk) = bv;
    __syncthreads();
    v16h aF0 = fragA_ptr(Ash + (wm * 32 + n) * 40, kh);
    v16h aF1 = fragA_ptr(Ash + (wm * 32 + 16 + n) * 40, kh);
    v16h bF0 = fragB_ptr(Bsh + (wn * 32 + n) * 40, kh);
    v16h bF1 = fragB_ptr(Bsh + (wn * 32 + 16 + n) * 40, kh);
    acc[0][0] = WMMA_F32(aF0, bF0, acc[0][0]);
    acc[0][1] = WMMA_F32(aF0, bF1, acc[0][1]);
    acc[1][0] = WMMA_F32(aF1, bF0, acc[1][0]);
    acc[1][1] = WMMA_F32(aF1, bF1, acc[1][1]);
    __syncthreads();
  }
#pragma unroll
  for (int i = 0; i < 2; ++i)
#pragma unroll
    for (int jn = 0; jn < 2; ++jn){
      int col = nBase + wn * 32 + jn * 16 + n;
      float bval = bias[col];
#pragma unroll
      for (int r = 0; r < 8; ++r){
        int row = mBase + wm * 32 + i * 16 + r + (mg << 3);
        int wi = row >> 7, tk = row & 127;
        int b = wi / NWIN_B, rem = wi % NWIN_B;
        int d  = ((rem >> 6) * 2 + (tk >> 6) + 1) % 6;
        int hh = ((((rem >> 3) & 7) << 3) + ((tk >> 3) & 7) + 4) & 63;
        int ww = (((rem & 7) << 3) + (tk & 7) + 4) & 63;
        long gi = (((((long)b * 6 + d) * 64 + hh) * 64 + ww) * CDIM) + col;
        X1[gi] = xin[gi] + acc[i][jn][r] + bval;
      }
    }
}

// fused GEGLU: Hmid = gelu(A@W11 + b1) * (A@W12 + b2)
__global__ __launch_bounds__(256) void k_gemm_geglu(const _Float16* __restrict__ A,
    const _Float16* __restrict__ B1, const _Float16* __restrict__ B2,
    const float* __restrict__ b1, const float* __restrict__ b2, _Float16* __restrict__ Out)
{
  const int K = CDIM, Nc = HID2;
  __shared__ _Float16 Ash[128 * 40];
  __shared__ _Float16 Bs1[64 * 40];
  __shared__ _Float16 Bs2[64 * 40];
  int t = threadIdx.x, lane = t & 31, wv = t >> 5;
  int wm = wv >> 1, wn = wv & 1;
  int n = lane & 15, kh = lane >> 4, mg = lane >> 4;
  int mBase = blockIdx.x * 128, nBase = blockIdx.y * 64;
  v8f accg[2][2] = {};
  v8f accl[2][2] = {};
  for (int k0 = 0; k0 < K; k0 += 32){
    int ar = t >> 1, ac = (t & 1) << 4;
    const uint4* as = reinterpret_cast<const uint4*>(A + (long)(mBase + ar) * K + k0 + ac);
    if (k0 + 32 < K) __builtin_prefetch(as + 4, 0, 0);
    uint4 a0 = as[0], a1 = as[1];
    uint4* ad = reinterpret_cast<uint4*>(Ash + ar * 40 + ac);
    ad[0] = a0; ad[1] = a1;
    int bn = t >> 2, bk = (t & 3) << 3;
    uint4 bv1 = *reinterpret_cast<const uint4*>(B1 + (long)(nBase + bn) * K + k0 + bk);
    uint4 bv2 = *reinterpret_cast<const uint4*>(B2 + (long)(nBase + bn) * K + k0 + bk);
    *reinterpret_cast<uint4*>(Bs1 + bn * 40 + bk) = bv1;
    *reinterpret_cast<uint4*>(Bs2 + bn * 40 + bk) = bv2;
    __syncthreads();
    v16h aF0 = fragA_ptr(Ash + (wm * 32 + n) * 40, kh);
    v16h aF1 = fragA_ptr(Ash + (wm * 32 + 16 + n) * 40, kh);
    v16h g0 = fragB_ptr(Bs1 + (wn * 32 + n) * 40, kh);
    v16h g1 = fragB_ptr(Bs1 + (wn * 32 + 16 + n) * 40, kh);
    v16h l0 = fragB_ptr(Bs2 + (wn * 32 + n) * 40, kh);
    v16h l1 = fragB_ptr(Bs2 + (wn * 32 + 16 + n) * 40, kh);
    accg[0][0] = WMMA_F32(aF0, g0, accg[0][0]);
    accg[0][1] = WMMA_F32(aF0, g1, accg[0][1]);
    accg[1][0] = WMMA_F32(aF1, g0, accg[1][0]);
    accg[1][1] = WMMA_F32(aF1, g1, accg[1][1]);
    accl[0][0] = WMMA_F32(aF0, l0, accl[0][0]);
    accl[0][1] = WMMA_F32(aF0, l1, accl[0][1]);
    accl[1][0] = WMMA_F32(aF1, l0, accl[1][0]);
    accl[1][1] = WMMA_F32(aF1, l1, accl[1][1]);
    __syncthreads();
  }
#pragma unroll
  for (int i = 0; i < 2; ++i)
#pragma unroll
    for (int jn = 0; jn < 2; ++jn){
      int col = nBase + wn * 32 + jn * 16 + n;
      float bv1 = b1[col], bv2 = b2[col];
#pragma unroll
      for (int r = 0; r < 8; ++r){
        int row = mBase + wm * 32 + i * 16 + r + (mg << 3);
        float v1 = accg[i][jn][r] + bv1;
        float v2 = accl[i][jn][r] + bv2;
        float g = 0.5f * v1 * (1.f + erff(v1 * 0.70710678118654752f));
        Out[(long)row * Nc + col] = (_Float16)(g * v2);
      }
    }
}

// fc2 GEMM with fused residual, writes final f32 output
__global__ __launch_bounds__(256) void k_gemm_fc2(const _Float16* __restrict__ A,
    const _Float16* __restrict__ Bw, const float* __restrict__ bias,
    const float* __restrict__ X1, float* __restrict__ Out)
{
  const int K = HID2, Nc = CDIM;
  __shared__ _Float16 Ash[128 * 40];
  __shared__ _Float16 Bsh[64 * 40];
  int t = threadIdx.x, lane = t & 31, wv = t >> 5;
  int wm = wv >> 1, wn = wv & 1;
  int n = lane & 15, kh = lane >> 4, mg = lane >> 4;
  int mBase = blockIdx.x * 128, nBase = blockIdx.y * 64;
  v8f acc[2][2] = {};
  for (int k0 = 0; k0 < K; k0 += 32){
    int ar = t >> 1, ac = (t & 1) << 4;
    const uint4* as = reinterpret_cast<const uint4*>(A + (long)(mBase + ar) * K + k0 + ac);
    if (k0 + 32 < K) __builtin_prefetch(as + 4, 0, 0);
    uint4 a0 = as[0], a1 = as[1];
    uint4* ad = reinterpret_cast<uint4*>(Ash + ar * 40 + ac);
    ad[0] = a0; ad[1] = a1;
    int bn = t >> 2, bk = (t & 3) << 3;
    uint4 bv = *reinterpret_cast<const uint4*>(Bw + (long)(nBase + bn) * K + k0 + bk);
    *reinterpret_cast<uint4*>(Bsh + bn * 40 + bk) = bv;
    __syncthreads();
    v16h aF0 = fragA_ptr(Ash + (wm * 32 + n) * 40, kh);
    v16h aF1 = fragA_ptr(Ash + (wm * 32 + 16 + n) * 40, kh);
    v16h bF0 = fragB_ptr(Bsh + (wn * 32 + n) * 40, kh);
    v16h bF1 = fragB_ptr(Bsh + (wn * 32 + 16 + n) * 40, kh);
    acc[0][0] = WMMA_F32(aF0, bF0, acc[0][0]);
    acc[0][1] = WMMA_F32(aF0, bF1, acc[0][1]);
    acc[1][0] = WMMA_F32(aF1, bF0, acc[1][0]);
    acc[1][1] = WMMA_F32(aF1, bF1, acc[1][1]);
    __syncthreads();
  }
#pragma unroll
  for (int i = 0; i < 2; ++i)
#pragma unroll
    for (int jn = 0; jn < 2; ++jn){
      int col = nBase + wn * 32 + jn * 16 + n;
      float bval = bias[col];
#pragma unroll
      for (int r = 0; r < 8; ++r){
        int row = mBase + wm * 32 + i * 16 + r + (mg << 3);
        long gi = (long)row * Nc + col;
        Out[gi] = X1[gi] + acc[i][jn][r] + bval;
      }
    }
}

// ============================ attention kernels ============================

// self attention: one block per (window, head); 8 waves x 16 rows
__global__ __launch_bounds__(256) void k_attn_self(const _Float16* __restrict__ qkv,
    const float* __restrict__ rpbd, const float* __restrict__ mask, _Float16* __restrict__ aout)
{
  __shared__ _Float16 Ksh[128 * 40];       // [token][hd]
  __shared__ _Float16 VshT[32 * 136];      // [hd][token]
  __shared__ _Float16 Psh[8][16 * 136];    // per-wave P rows
  int wi = blockIdx.x / NHEAD, h = blockIdx.x % NHEAD;
  int t = threadIdx.x, lane = t & 31, wv = t >> 5;
  {
    int tok = t >> 1, kb = (t & 1) << 4;
    long rowoff = (long)(wi * NTOK + tok) * QKVC;
    const uint4* ks = reinterpret_cast<const uint4*>(qkv + rowoff + CDIM + h * HDIM + kb);
    uint4 k0 = ks[0], k1 = ks[1];
    uint4* kd = reinterpret_cast<uint4*>(Ksh + tok * 40 + kb);
    kd[0] = k0; kd[1] = k1;
    v8h va = *reinterpret_cast<const v8h*>(qkv + rowoff + 2 * CDIM + h * HDIM + kb);
    v8h vb = *reinterpret_cast<const v8h*>(qkv + rowoff + 2 * CDIM + h * HDIM + kb + 8);
#pragma unroll
    for (int e = 0; e < 8; ++e){
      VshT[(kb + e) * 136 + tok]     = va[e];
      VshT[(kb + e + 8) * 136 + tok] = vb[e];
    }
  }
  __syncthreads();
  int rt = wv, n = lane & 15, mg = lane >> 4, kh = lane >> 4;
  v16h qF = fragA_ptr(qkv + (long)(wi * NTOK + rt * 16 + n) * QKVC + h * HDIM, kh);
  v8f S[8];
#pragma unroll
  for (int ct = 0; ct < 8; ++ct){
    v16h bF = fragB_ptr(Ksh + (ct * 16 + n) * 40, kh);
    v8f z = {};
    S[ct] = WMMA_F32(qF, bF, z);
  }
  const float* rpbh = rpbd + h * (NTOK * NTOK);
  const float* mw = mask + (long)(wi % NWIN_B) * (NTOK * NTOK);
  float rmax[8], rsum[8];
#pragma unroll
  for (int r = 0; r < 8; ++r) rmax[r] = -3.0e30f;
#pragma unroll
  for (int ct = 0; ct < 8; ++ct){
    int colg = ct * 16 + n;
#pragma unroll
    for (int r = 0; r < 8; ++r){
      int rowg = rt * 16 + r + (mg << 3);
      float s = S[ct][r] + rpbh[rowg * NTOK + colg] + mw[rowg * NTOK + colg];
      S[ct][r] = s;
      rmax[r] = fmaxf(rmax[r], s);
    }
  }
#pragma unroll
  for (int m = 1; m < 16; m <<= 1)
#pragma unroll
    for (int r = 0; r < 8; ++r) rmax[r] = fmaxf(rmax[r], __shfl_xor(rmax[r], m));
#pragma unroll
  for (int r = 0; r < 8; ++r) rsum[r] = 0.f;
#pragma unroll
  for (int ct = 0; ct < 8; ++ct)
#pragma unroll
    for (int r = 0; r < 8; ++r){
      float e = __expf(S[ct][r] - rmax[r]);
      S[ct][r] = e; rsum[r] += e;
    }
#pragma unroll
  for (int m = 1; m < 16; m <<= 1)
#pragma unroll
    for (int r = 0; r < 8; ++r) rsum[r] += __shfl_xor(rsum[r], m);
  _Float16* pw = Psh[rt];
#pragma unroll
  for (int ct = 0; ct < 8; ++ct)
#pragma unroll
    for (int r = 0; r < 8; ++r)
      pw[(r + (mg << 3)) * 136 + ct * 16 + n] = (_Float16)(S[ct][r] / rsum[r]);
  __syncthreads();
#pragma unroll
  for (int n2 = 0; n2 < 2; ++n2){
    v8f o = {};
#pragma unroll
    for (int kc = 0; kc < 4; ++kc){
      v16h aF = fragA_ptr(pw + n * 136 + kc * 32, kh);
      v16h bF = fragB_ptr(VshT + (n2 * 16 + n) * 136 + kc * 32, kh);
      o = WMMA_F32(aF, bF, o);
    }
#pragma unroll
    for (int r = 0; r < 8; ++r){
      long rowg = (long)wi * NTOK + rt * 16 + r + (mg << 3);
      int col = CDIM + h * HDIM + n2 * 16 + n;
      aout[rowg * (2 * CDIM) + col] = (_Float16)o[r];
    }
  }
}

// mutual attention: waves 0-3 -> x1 = attn(q2,k1,v1); waves 4-7 -> x2 = attn(q1,k2,v2)
__global__ __launch_bounds__(256) void k_attn_mut(const _Float16* __restrict__ qkv,
    const float* __restrict__ mask, _Float16* __restrict__ aout)
{
  __shared__ _Float16 Ksh[128 * 40];
  __shared__ _Float16 VshT[32 * 136];
  __shared__ _Float16 Psh[8][16 * 72];
  int wi = blockIdx.x / NHEAD, h = blockIdx.x % NHEAD;
  int t = threadIdx.x, lane = t & 31, wv = t >> 5;
  {
    int tok = t >> 1, kb = (t & 1) << 4;
    long rowoff = (long)(wi * NTOK + tok) * QKVC;
    const uint4* ks = reinterpret_cast<const uint4*>(qkv + rowoff + CDIM + h * HDIM + kb);
    uint4 k0 = ks[0], k1 = ks[1];
    uint4* kd = reinterpret_cast<uint4*>(Ksh + tok * 40 + kb);
    kd[0] = k0; kd[1] = k1;
    v8h va = *reinterpret_cast<const v8h*>(qkv + rowoff + 2 * CDIM + h * HDIM + kb);
    v8h vb = *reinterpret_cast<const v8h*>(qkv + rowoff + 2 * CDIM + h * HDIM + kb + 8);
#pragma unroll
    for (int e = 0; e < 8; ++e){
      VshT[(kb + e) * 136 + tok]     = va[e];
      VshT[(kb + e + 8) * 136 + tok] = vb[e];
    }
  }
  __syncthreads();
  int task = wv >> 2, rtl = wv & 3;
  int qbase  = (task == 0) ? 64 : 0;
  int kvbase = (task == 0) ? 0 : 64;
  int n = lane & 15, mg = lane >> 4, kh = lane >> 4;
  v16h qF = fragA_ptr(qkv + (long)(wi * NTOK + qbase + rtl * 16 + n) * QKVC + h * HDIM, kh);
  v8f S[4];
#pragma unroll
  for (int ct = 0; ct < 4; ++ct){
    v16h bF = fragB_ptr(Ksh + (kvbase + ct * 16 + n) * 40, kh);
    v8f z = {};
    S[ct] = WMMA_F32(qF, bF, z);
  }
  const float* mw = mask + (long)(wi % NWIN_B) * (NTOK * NTOK);
  float rmax[8], rsum[8];
#pragma unroll
  for (int r = 0; r < 8; ++r) rmax[r] = -3.0e30f;
#pragma unroll
  for (int ct = 0; ct < 4; ++ct){
    int colg = ct * 16 + n;
#pragma unroll
    for (int r = 0; r < 8; ++r){
      int rowg = rtl * 16 + r + (mg << 3);
      float s = S[ct][r] + mw[rowg * NTOK + colg];   // mask[:, :64, :64]
      S[ct][r] = s;
      rmax[r] = fmaxf(rmax[r], s);
    }
  }
#pragma unroll
  for (int m = 1; m < 16; m <<= 1)
#pragma unroll
    for (int r = 0; r < 8; ++r) rmax[r] = fmaxf(rmax[r], __shfl_xor(rmax[r], m));
#pragma unroll
  for (int r = 0; r < 8; ++r) rsum[r] = 0.f;
#pragma unroll
  for (int ct = 0; ct < 4; ++ct)
#pragma unroll
    for (int r = 0; r < 8; ++r){
      float e = __expf(S[ct][r] - rmax[r]);
      S[ct][r] = e; rsum[r] += e;
    }
#pragma unroll
  for (int m = 1; m < 16; m <<= 1)
#pragma unroll
    for (int r = 0; r < 8; ++r) rsum[r] += __shfl_xor(rsum[r], m);
  _Float16* pw = Psh[wv];
#pragma unroll
  for (int ct = 0; ct < 4; ++ct)
#pragma unroll
    for (int r = 0; r < 8; ++r)
      pw[(r + (mg << 3)) * 72 + ct * 16 + n] = (_Float16)(S[ct][r] / rsum[r]);
  __syncthreads();
#pragma unroll
  for (int n2 = 0; n2 < 2; ++n2){
    v8f o = {};
#pragma unroll
    for (int kc = 0; kc < 2; ++kc){
      v16h aF = fragA_ptr(pw + n * 72 + kc * 32, kh);
      v16h bF = fragB_ptr(VshT + (n2 * 16 + n) * 136 + kvbase + kc * 32, kh);
      o = WMMA_F32(aF, bF, o);
    }
#pragma unroll
    for (int r = 0; r < 8; ++r){
      long rowg = (long)wi * NTOK + task * 64 + rtl * 16 + r + (mg << 3);
      int col = h * HDIM + n2 * 16 + n;
      aout[rowg * (2 * CDIM) + col] = (_Float16)o[r];
    }
  }
}

// ============================ host launcher ============================

extern "C" void kernel_launch(void* const* d_in, const int* in_sizes, int n_in,
                              void* d_out, int out_size, void* d_ws, size_t ws_size,
                              hipStream_t stream)
{
  (void)in_sizes; (void)n_in; (void)out_size; (void)ws_size;
  const float* x     = (const float*)d_in[0];
  const float* mask  = (const float*)d_in[1];
  const int*   rpi   = (const int*)d_in[2];
  const float* pos   = (const float*)d_in[3];
  const float* n1w   = (const float*)d_in[4];
  const float* n1b   = (const float*)d_in[5];
  const float* qkvsW = (const float*)d_in[6];
  const float* qkvsB = (const float*)d_in[7];
  const float* qkvmW = (const float*)d_in[8];
  const float* qkvmB = (const float*)d_in[9];
  const float* rtab  = (const float*)d_in[10];
  const float* projW = (const float*)d_in[11];
  const float* projB = (const float*)d_in[12];
  const float* n2w   = (const float*)d_in[13];
  const float* n2b   = (const float*)d_in[14];
  const float* fc11W = (const float*)d_in[15];
  const float* fc11B = (const float*)d_in[16];
  const float* fc12W = (const float*)d_in[17];
  const float* fc12B = (const float*)d_in[18];
  const float* fc2W  = (const float*)d_in[19];
  const float* fc2B  = (const float*)d_in[20];

  char* ws = (char*)d_ws;
  size_t off = 0;
  auto alloc = [&](size_t bytes) -> void* {
    void* p = ws + off; off += (bytes + 255) & ~(size_t)255; return p;
  };
  _Float16* XW    = (_Float16*)alloc((size_t)TTOK * CDIM * 2);
  _Float16* XWPB  = (_Float16*)alloc((size_t)TTOK * CDIM * 2);
  _Float16* QKV   = (_Float16*)alloc((size_t)TTOK * QKVC * 2);
  _Float16* AOUT  = (_Float16*)alloc((size_t)TTOK * 2 * CDIM * 2);
  float*    X1    = (float*)alloc((size_t)TTOK * CDIM * 4);
  float*    RPBD  = (float*)alloc((size_t)NHEAD * NTOK * NTOK * 4);
  _Float16* WqkvS = (_Float16*)alloc((size_t)QKVC * CDIM * 2);
  _Float16* WqkvM = (_Float16*)alloc((size_t)QKVC * CDIM * 2);
  _Float16* Wproj = (_Float16*)alloc((size_t)2 * CDIM * CDIM * 2);
  _Float16* W11   = (_Float16*)alloc((size_t)HID2 * CDIM * 2);
  _Float16* W12   = (_Float16*)alloc((size_t)HID2 * CDIM * 2);
  _Float16* W2    = (_Float16*)alloc((size_t)CDIM * HID2 * 2);
  _Float16* H2    = XW;    // reuse (XW dead after qkv_self GEMM)
  _Float16* HMID  = QKV;   // reuse (QKV dead after mutual attention)

  const float scale = 0.17677669529663687f;  // hd^-0.5
  auto cg = [](int nel){ return (nel + 255) / 256; };

  k_conv<<<cg(QKVC * CDIM), 256, 0, stream>>>(qkvsW, WqkvS, QKVC * CDIM);
  k_conv<<<cg(QKVC * CDIM), 256, 0, stream>>>(qkvmW, WqkvM, QKVC * CDIM);
  k_conv<<<cg(CDIM * 2 * CDIM), 256, 0, stream>>>(projW, Wproj, CDIM * 2 * CDIM);
  k_conv<<<cg(HID2 * CDIM), 256, 0, stream>>>(fc11W, W11, HID2 * CDIM);
  k_conv<<<cg(HID2 * CDIM), 256, 0, stream>>>(fc12W, W12, HID2 * CDIM);
  k_conv<<<cg(CDIM * HID2), 256, 0, stream>>>(fc2W, W2, CDIM * HID2);
  k_rpb<<<(NHEAD * NTOK * NTOK) / 256, 256, 0, stream>>>(rtab, rpi, RPBD);
  k_prep1<<<TTOK / 8, 256, 0, stream>>>(x, pos, n1w, n1b, XW, XWPB);
  k_gemm_qkv<<<dim3(TTOK / 128, QKVC / 64), 256, 0, stream>>>(XW, WqkvS, qkvsB, QKV, CDIM, QKVC, scale);
  k_attn_self<<<BWIN * NHEAD, 256, 0, stream>>>(QKV, RPBD, mask, AOUT);
  k_gemm_qkv<<<dim3(TTOK / 128, QKVC / 64), 256, 0, stream>>>(XWPB, WqkvM, qkvmB, QKV, CDIM, QKVC, scale);
  k_attn_mut<<<BWIN * NHEAD, 256, 0, stream>>>(QKV, mask, AOUT);
  k_gemm_proj<<<dim3(TTOK / 128, CDIM / 64), 256, 0, stream>>>(AOUT, Wproj, projB, x, X1);
  k_prep2<<<TTOK / 8, 256, 0, stream>>>(X1, n2w, n2b, H2);
  k_gemm_geglu<<<dim3(TTOK / 128, HID2 / 64), 256, 0, stream>>>(H2, W11, W12, fc11B, fc12B, HMID);
  k_gemm_fc2<<<dim3(TTOK / 128, CDIM / 64), 256, 0, stream>>>(HMID, W2, fc2B, X1, (float*)d_out);
}
